// CausalSelfAttention_35931696398729
// MI455X (gfx1250) — matrix operand order repaired and checked by reading, not compiled, not run on hardware
//
#include <hip/hip_runtime.h>

typedef __attribute__((ext_vector_type(16))) _Float16 v16h;
typedef __attribute__((ext_vector_type(8)))  float    v8f;
typedef __attribute__((ext_vector_type(4)))  int      v4i;

// ---------------- problem constants ----------------
constexpr int B_    = 2;
constexpr int T_    = 2048;
constexpr int D_    = 2048;
constexpr int NH_   = 16;
constexpr int NKV_  = 4;
constexpr int HD_   = 128;
constexpr int LDQ_  = NH_ * HD_;   // 2048
constexpr int LDKV_ = NKV_ * HD_;  // 512
constexpr int BT_   = B_ * T_;     // 4096

// ---------------- workspace layout (bytes) ----------------
constexpr size_t OFF_X16  = 0;
constexpr size_t OFF_WQ16 = OFF_X16  + (size_t)BT_ * D_ * 2;
constexpr size_t OFF_WK16 = OFF_WQ16 + (size_t)D_ * LDQ_ * 2;
constexpr size_t OFF_WV16 = OFF_WK16 + (size_t)D_ * LDKV_ * 2;
constexpr size_t OFF_WO16 = OFF_WV16 + (size_t)D_ * LDKV_ * 2;
constexpr size_t OFF_Q16  = OFF_WO16 + (size_t)D_ * D_ * 2;
constexpr size_t OFF_K16  = OFF_Q16  + (size_t)BT_ * LDQ_ * 2;
constexpr size_t OFF_V16  = OFF_K16  + (size_t)BT_ * LDKV_ * 2;
constexpr size_t OFF_A16  = OFF_V16  + (size_t)BT_ * LDKV_ * 2;
constexpr size_t OFF_QF   = OFF_A16  + (size_t)BT_ * D_ * 2;
constexpr size_t OFF_KF   = OFF_QF   + (size_t)BT_ * LDQ_ * 4;
constexpr size_t OFF_VF   = OFF_KF   + (size_t)BT_ * LDKV_ * 4;

// ---------------- CDNA5 async global->LDS (guarded probe) ----------------
#if defined(__has_builtin)
#if __has_builtin(__builtin_amdgcn_global_load_async_to_lds_b128)
#define USE_ASYNC_LDS 1
#endif
#if __has_builtin(__builtin_amdgcn_s_wait_asynccnt)
#define WAIT_ASYNC() __builtin_amdgcn_s_wait_asynccnt(0)
#endif
#endif
#ifndef USE_ASYNC_LDS
#define USE_ASYNC_LDS 0
#endif
#ifndef WAIT_ASYNC
#define WAIT_ASYNC() asm volatile("s_wait_asynccnt 0x0" ::: "memory")
#endif

#if USE_ASYNC_LDS
typedef __attribute__((address_space(1))) v4i* gv4p;
typedef __attribute__((address_space(3))) v4i* lv4p;
__device__ __forceinline__ void async_copy_b128(const _Float16* g, _Float16* l) {
  __builtin_amdgcn_global_load_async_to_lds_b128(
      (gv4p)(void*)g, (lv4p)(void*)l, 0, 0);
}
#endif

// ---------------- helpers ----------------
__device__ __forceinline__ v16h pack2(const _Float16* p0, const _Float16* p1) {
  union { uint4 u[2]; v16h h; } z;
  z.u[0] = *reinterpret_cast<const uint4*>(p0);
  z.u[1] = *reinterpret_cast<const uint4*>(p1);
  return z.h;
}

__device__ __forceinline__ v8f wmma16(v16h a, v16h b, v8f c) {
  return __builtin_amdgcn_wmma_f32_16x16x32_f16(false, a, false, b, (short)0, c, false, false);
}

// ---------------- fp32 -> f16 convert (vectorized x4) ----------------
__global__ void cvt_f32_f16(const float* __restrict__ in, _Float16* __restrict__ out, int n4) {
  int i = blockIdx.x * blockDim.x + threadIdx.x;
  int stride = gridDim.x * blockDim.x;
  const float4* in4 = reinterpret_cast<const float4*>(in);
  uint2* out4 = reinterpret_cast<uint2*>(out);
  for (; i < n4; i += stride) {
    float4 f = in4[i];
    union { uint2 u; _Float16 h[4]; } z;
    z.h[0] = (_Float16)f.x; z.h[1] = (_Float16)f.y;
    z.h[2] = (_Float16)f.z; z.h[3] = (_Float16)f.w;
    out4[i] = z.u;
  }
}

// ---------------- tiled WMMA GEMM: C[M,N] (f32) = A16[M,K] @ B16[K,N] ----------------
// 256 threads = 8 waves; block tile 128x128; wave tile 64x32; K step 32; double-buffered LDS.
__global__ __launch_bounds__(256) void gemm_f16f32(const _Float16* __restrict__ A,
                                                   const _Float16* __restrict__ Bm,
                                                   float* __restrict__ C,
                                                   int M, int N, int K) {
  __shared__ alignas(16) _Float16 As[2][128][32];   // row-major [m][k]
  __shared__ alignas(16) _Float16 Bt[2][128][32];   // transposed [n][k]
  const int tid  = threadIdx.x;
  const int lane = tid & 31;
  const int w    = tid >> 5;
  const int wm   = w & 1;
  const int wn   = w >> 1;
  const int grp  = lane >> 4;
  const int l16  = lane & 15;
  const int bm   = blockIdx.y * 128;
  const int bn   = blockIdx.x * 128;
  // staging coordinates
  const int am  = tid >> 1, ako = (tid & 1) * 16;          // A: 1 row x 16 halfs
  const int bkq = (tid & 7) * 4, bn4 = (tid >> 3) * 4;     // B: 4 k-rows x 4 n-cols

  v8f zero = {};
  v8f acc[4][2];
#pragma unroll
  for (int i = 0; i < 4; i++)
#pragma unroll
    for (int j = 0; j < 2; j++) acc[i][j] = zero;

  union RowU { uint2 u; _Float16 h[4]; };

  // ---- prologue: stage tile 0 into buffer 0 ----
  {
#if USE_ASYNC_LDS
    const _Float16* g = A + (size_t)(bm + am) * K + ako;
    async_copy_b128(g,     &As[0][am][ako]);
    async_copy_b128(g + 8, &As[0][am][ako + 8]);
#else
    const uint4* src = reinterpret_cast<const uint4*>(A + (size_t)(bm + am) * K + ako);
    uint4 a0 = src[0], a1 = src[1];
    uint4* dst = reinterpret_cast<uint4*>(&As[0][am][ako]);
    dst[0] = a0; dst[1] = a1;
#endif
    RowU rows[4];
#pragma unroll
    for (int i = 0; i < 4; i++)
      rows[i].u = *reinterpret_cast<const uint2*>(Bm + (size_t)(bkq + i) * N + bn + bn4);
#pragma unroll
    for (int j = 0; j < 4; j++) {
      RowU col;
#pragma unroll
      for (int i = 0; i < 4; i++) col.h[i] = rows[i].h[j];
      *reinterpret_cast<uint2*>(&Bt[0][bn4 + j][bkq]) = col.u;
    }
#if USE_ASYNC_LDS
    WAIT_ASYNC();
#endif
  }
  __syncthreads();

  const int nsteps = K / 32;
  for (int s = 0; s < nsteps; s++) {
    const int cur = s & 1, nxt = cur ^ 1;
    const int k1 = (s + 1) * 32;
    const bool more = (s + 1) < nsteps;

    // issue next-tile global traffic early (overlaps with WMMA below)
    RowU rows[4];
#if !USE_ASYNC_LDS
    uint4 na0, na1;
#endif
    if (more) {
#if USE_ASYNC_LDS
      const _Float16* g = A + (size_t)(bm + am) * K + k1 + ako;
      async_copy_b128(g,     &As[nxt][am][ako]);
      async_copy_b128(g + 8, &As[nxt][am][ako + 8]);
#else
      const uint4* src = reinterpret_cast<const uint4*>(A + (size_t)(bm + am) * K + k1 + ako);
      na0 = src[0]; na1 = src[1];
#endif
#pragma unroll
      for (int i = 0; i < 4; i++)
        rows[i].u = *reinterpret_cast<const uint2*>(Bm + (size_t)(k1 + bkq + i) * N + bn + bn4);
    }

    // ---- compute on buffer `cur` ----
    v16h bf[2];
#pragma unroll
    for (int nt = 0; nt < 2; nt++) {
      const _Float16* p = &Bt[cur][wn * 32 + nt * 16 + l16][grp * 8];
      bf[nt] = pack2(p, p + 16);
    }
#pragma unroll
    for (int mt = 0; mt < 4; mt++) {
      const _Float16* p = &As[cur][wm * 64 + mt * 16 + l16][grp * 8];
      v16h af = pack2(p, p + 16);
#pragma unroll
      for (int nt = 0; nt < 2; nt++) acc[mt][nt] = wmma16(af, bf[nt], acc[mt][nt]);
    }

    if (more) {
#if !USE_ASYNC_LDS
      uint4* dst = reinterpret_cast<uint4*>(&As[nxt][am][ako]);
      dst[0] = na0; dst[1] = na1;
#endif
#pragma unroll
      for (int j = 0; j < 4; j++) {
        RowU col;
#pragma unroll
        for (int i = 0; i < 4; i++) col.h[i] = rows[i].h[j];
        *reinterpret_cast<uint2*>(&Bt[nxt][bn4 + j][bkq]) = col.u;
      }
#if USE_ASYNC_LDS
      WAIT_ASYNC();
#endif
      __syncthreads();
    }
  }

  // ---- epilogue: store C ----
#pragma unroll
  for (int mt = 0; mt < 4; mt++)
#pragma unroll
    for (int nt = 0; nt < 2; nt++) {
      int n = bn + wn * 32 + nt * 16 + l16;
#pragma unroll
      for (int r = 0; r < 8; r++) {
        int m = bm + wm * 64 + mt * 16 + r + 8 * grp;
        C[(size_t)m * N + n] = acc[mt][nt][r];
      }
    }
}

// ---------------- RoPE + RMSNorm (fp32 in, f16 out) ----------------
__global__ __launch_bounds__(128) void rope_rmsnorm(const float* __restrict__ in,
                                                    const float* __restrict__ gamma,
                                                    _Float16* __restrict__ out,
                                                    int H, const int* __restrict__ posp) {
  const int token = blockIdx.x;
  const int h     = blockIdx.y;
  const int d     = threadIdx.x;
  const int t     = token & (T_ - 1);
  const int pos   = *posp;
  __shared__ float buf[HD_];
  __shared__ float wsum[4];
  const size_t off = (size_t)token * H * HD_ + (size_t)h * HD_;

  if (d < 64) {
    float ef   = (float)d * (1.0f / 64.0f);
    float invf = __expf(-ef * 9.2103403719761836f);  // ln(10000)
    float ang  = (float)(pos + t) * invf;
    float c = __cosf(ang), s = __sinf(ang);
    float t1 = in[off + d], t2 = in[off + d + 64];
    buf[d]      = t1 * c - t2 * s;
    buf[d + 64] = t1 * s + t2 * c;
  }
  __syncthreads();
  float v  = buf[d];
  float ss = v * v;
#pragma unroll
  for (int o = 16; o > 0; o >>= 1) ss += __shfl_xor(ss, o, 32);
  if ((d & 31) == 0) wsum[d >> 5] = ss;
  __syncthreads();
  float tot = wsum[0] + wsum[1] + wsum[2] + wsum[3];
  float r   = rsqrtf(tot * (1.0f / 128.0f) + 1e-6f);
  out[off + d] = (_Float16)(v * r * gamma[d]);
}

// ---------------- flash attention (causal, GQA) ----------------
// grid (T/64, NH, B), block 128 = 4 waves; wave owns 16 q rows; kv tiles of 32; V double-buffered.
__global__ __launch_bounds__(128) void flash_attn(const _Float16* __restrict__ q,
                                                  const _Float16* __restrict__ k,
                                                  const _Float16* __restrict__ v,
                                                  _Float16* __restrict__ o_out) {
  __shared__ alignas(16) _Float16 Vt[2][128][32];    // V tile transposed [hd][kv]
  __shared__ alignas(16) _Float16 Pbuf[4][16][32];   // per-wave probs [m][kv]
  const int tid  = threadIdx.x;
  const int lane = tid & 31;
  const int w    = tid >> 5;
  const int grp  = lane >> 4;
  const int l16  = lane & 15;
  const int qt   = blockIdx.x;
  const int h    = blockIdx.y;
  const int bb   = blockIdx.z;
  const int kvh  = h >> 2;                // G = 4
  const int q0   = qt * 64;
  const int mrow = q0 + w * 16;
  const float scale = 0.08838834764831845f;  // 1/sqrt(128)
  // V staging coords: 4 kv-rows x 8 hd per thread
  const int vs4 = (tid & 7) * 4, vhd8 = (tid >> 3) * 8;

  // Q fragments, held for whole kernel: 16 rows x 128 hd
  v16h qf[4];
  {
    const int m = mrow + l16;
    const _Float16* qrow = q + ((size_t)(bb * T_ + m)) * LDQ_ + h * HD_;
#pragma unroll
    for (int kc = 0; kc < 4; kc++) {
      const _Float16* p = qrow + kc * 32 + grp * 8;
      qf[kc] = pack2(p, p + 16);
    }
  }

  v8f zero = {};
  v8f o[8];
#pragma unroll
  for (int i = 0; i < 8; i++) o[i] = zero;
  float Mr[8], Lr[8];
#pragma unroll
  for (int r = 0; r < 8; r++) { Mr[r] = -1e30f; Lr[r] = 0.0f; }

  union RowU { uint4 u; _Float16 h[8]; };

  // prologue: stage V tile 0 into buffer 0
  {
    RowU rows[4];
#pragma unroll
    for (int i = 0; i < 4; i++)
      rows[i].u = *reinterpret_cast<const uint4*>(
          v + ((size_t)(bb * T_ + vs4 + i)) * LDKV_ + kvh * HD_ + vhd8);
#pragma unroll
    for (int j = 0; j < 8; j++) {
      union { uint2 u; _Float16 h[4]; } col;
#pragma unroll
      for (int i = 0; i < 4; i++) col.h[i] = rows[i].h[j];
      *reinterpret_cast<uint2*>(&Vt[0][vhd8 + j][vs4]) = col.u;
    }
  }
  __syncthreads();

  const int ntiles = (q0 + 64) / 32;
  for (int idx = 0; idx < ntiles; idx++) {
    const int s0  = idx * 32;
    const int cur = idx & 1, nxt = cur ^ 1;
    const bool more = (idx + 1) < ntiles;

    // issue next V tile global loads + prefetch next K tile (overlap with WMMA)
    RowU vrows[4];
    if (more) {
      const int s0n = s0 + 32;
#pragma unroll
      for (int i = 0; i < 4; i++)
        vrows[i].u = *reinterpret_cast<const uint4*>(
            v + ((size_t)(bb * T_ + s0n + vs4 + i)) * LDKV_ + kvh * HD_ + vhd8);
      __builtin_prefetch(k + ((size_t)(bb * T_ + s0n + (tid & 31))) * LDKV_ + kvh * HD_, 0, 0);
    }

    // ---- scores: S(16x32) = Q(16x128) @ K^T(128x32) ----
    v8f sa = zero, sb = zero;
#pragma unroll
    for (int kc = 0; kc < 4; kc++) {
      const _Float16* kb0 = k + ((size_t)(bb * T_ + s0 + l16)) * LDKV_ + kvh * HD_ + kc * 32 + grp * 8;
      const _Float16* kb1 = k + ((size_t)(bb * T_ + s0 + 16 + l16)) * LDKV_ + kvh * HD_ + kc * 32 + grp * 8;
      v16h b0 = pack2(kb0, kb0 + 16);
      v16h b1 = pack2(kb1, kb1 + 16);
      sa = wmma16(qf[kc], b0, sa);
      sb = wmma16(qf[kc], b1, sb);
    }

    // ---- online softmax per q row (row m = r + 8*grp in wave tile) ----
#pragma unroll
    for (int r = 0; r < 8; r++) {
      int mg  = mrow + r + 8 * grp;
      int sg0 = s0 + l16;
      int sg1 = s0 + 16 + l16;
      float a0 = sa[r] * scale; a0 = (sg0 > mg) ? -1e30f : a0;
      float a1 = sb[r] * scale; a1 = (sg1 > mg) ? -1e30f : a1;
      float rmax = fmaxf(a0, a1);
#pragma unroll
      for (int off = 8; off > 0; off >>= 1) rmax = fmaxf(rmax, __shfl_xor(rmax, off, 32));
      float nM   = fmaxf(Mr[r], rmax);
      float corr = __expf(Mr[r] - nM);
      float p0   = __expf(a0 - nM);
      float p1   = __expf(a1 - nM);
      float ps   = p0 + p1;
#pragma unroll
      for (int off = 8; off > 0; off >>= 1) ps += __shfl_xor(ps, off, 32);
      Lr[r] = Lr[r] * corr + ps;
      Mr[r] = nM;
#pragma unroll
      for (int t2 = 0; t2 < 8; t2++) o[t2][r] *= corr;
      Pbuf[w][r + 8 * grp][l16]      = (_Float16)p0;
      Pbuf[w][r + 8 * grp][16 + l16] = (_Float16)p1;
    }

    // ---- O += P(16x32) @ V(32x128) ----  (Pbuf is wave-private: DS in-order)
    v16h pa;
    { const _Float16* p = &Pbuf[w][l16][grp * 8]; pa = pack2(p, p + 16); }
#pragma unroll
    for (int t2 = 0; t2 < 8; t2++) {
      const _Float16* p = &Vt[cur][t2 * 16 + l16][grp * 8];
      v16h bv = pack2(p, p + 16);
      o[t2] = wmma16(pa, bv, o[t2]);
    }

    if (more) {
#pragma unroll
      for (int j = 0; j < 8; j++) {
        union { uint2 u; _Float16 h[4]; } col;
#pragma unroll
        for (int i = 0; i < 4; i++) col.h[i] = vrows[i].h[j];
        *reinterpret_cast<uint2*>(&Vt[nxt][vhd8 + j][vs4]) = col.u;
      }
      __syncthreads();
    }
  }

  // ---- normalize + write attention output (f16) ----
#pragma unroll
  for (int r = 0; r < 8; r++) {
    float inv = 1.0f / Lr[r];
    int m = mrow + r + 8 * grp;
    size_t base = ((size_t)(bb * T_ + m)) * (size_t)D_ + (size_t)h * HD_;
#pragma unroll
    for (int t2 = 0; t2 < 8; t2++)
      o_out[base + t2 * 16 + l16] = (_Float16)(o[t2][r] * inv);
  }
}

// ---------------- host launcher ----------------
extern "C" void kernel_launch(void* const* d_in, const int* in_sizes, int n_in,
                              void* d_out, int out_size, void* d_ws, size_t ws_size,
                              hipStream_t stream) {
  const float* x  = (const float*)d_in[0];
  const float* Wq = (const float*)d_in[1];
  const float* Wk = (const float*)d_in[2];
  const float* Wv = (const float*)d_in[3];
  const float* Wo = (const float*)d_in[4];
  const float* qg = (const float*)d_in[5];
  const float* kg = (const float*)d_in[6];
  const int*  pos = (const int*)d_in[7];

  char* ws = (char*)d_ws;
  _Float16* x16  = (_Float16*)(ws + OFF_X16);
  _Float16* Wq16 = (_Float16*)(ws + OFF_WQ16);
  _Float16* Wk16 = (_Float16*)(ws + OFF_WK16);
  _Float16* Wv16 = (_Float16*)(ws + OFF_WV16);
  _Float16* Wo16 = (_Float16*)(ws + OFF_WO16);
  _Float16* q16  = (_Float16*)(ws + OFF_Q16);
  _Float16* k16  = (_Float16*)(ws + OFF_K16);
  _Float16* v16  = (_Float16*)(ws + OFF_V16);
  _Float16* a16  = (_Float16*)(ws + OFF_A16);
  float*    qf   = (float*)(ws + OFF_QF);
  float*    kf   = (float*)(ws + OFF_KF);
  float*    vf   = (float*)(ws + OFF_VF);

  // 1) down-convert activations + weights to f16
  cvt_f32_f16<<<2048, 256, 0, stream>>>(x,  x16,  BT_ * D_ / 4);
  cvt_f32_f16<<<2048, 256, 0, stream>>>(Wq, Wq16, D_ * LDQ_ / 4);
  cvt_f32_f16<<<1024, 256, 0, stream>>>(Wk, Wk16, D_ * LDKV_ / 4);
  cvt_f32_f16<<<1024, 256, 0, stream>>>(Wv, Wv16, D_ * LDKV_ / 4);
  cvt_f32_f16<<<2048, 256, 0, stream>>>(Wo, Wo16, D_ * D_ / 4);

  // 2) QKV projections (WMMA GEMMs, fp32 accumulate/output)
  gemm_f16f32<<<dim3(LDQ_ / 128,  BT_ / 128), 256, 0, stream>>>(x16, Wq16, qf, BT_, LDQ_,  D_);
  gemm_f16f32<<<dim3(LDKV_ / 128, BT_ / 128), 256, 0, stream>>>(x16, Wk16, kf, BT_, LDKV_, D_);
  gemm_f16f32<<<dim3(LDKV_ / 128, BT_ / 128), 256, 0, stream>>>(x16, Wv16, vf, BT_, LDKV_, D_);

  // 3) RoPE + RMSNorm on q,k (fp32 math), emit f16; v just converted
  rope_rmsnorm<<<dim3(BT_, NH_),  128, 0, stream>>>(qf, qg, q16, NH_,  pos);
  rope_rmsnorm<<<dim3(BT_, NKV_), 128, 0, stream>>>(kf, kg, k16, NKV_, pos);
  cvt_f32_f16<<<1024, 256, 0, stream>>>(vf, v16, BT_ * LDKV_ / 4);

  // 4) causal GQA flash attention (WMMA for QK^T and PV)
  flash_attn<<<dim3(T_ / 64, NH_, B_), 128, 0, stream>>>(q16, k16, v16, a16);

  // 5) output projection -> fp32 result
  gemm_f16f32<<<dim3(D_ / 128, BT_ / 128), 256, 0, stream>>>(a16, Wo16, (float*)d_out, BT_, D_, D_);
}
